// ImprovedAttentionEdgeEmb_34256659153217
// MI455X (gfx1250) — matrix operand order repaired and checked by reading, not compiled, not run on hardware
//
#include <hip/hip_runtime.h>

typedef __attribute__((ext_vector_type(16))) _Float16 v16h;
typedef __attribute__((ext_vector_type(8)))  float    v8f;

#define CH       128
#define N_EDGES  625000
#define N_TILES  ((N_EDGES + 15) / 16)   // 39063 M-tiles of 16 edges
#define WPB      4                        // waves per block
#define BLOCK_T  (WPB * 32)

// ---------------------------------------------------------------------------
// prep_B: a = softmax(attn); W'[o,d] = W[o,d] * a[d]; emit f16 B fragments in
// contiguous-K B order (A matches):  fragment (nt,kt) -> 32 lanes x 16 halfs contiguous.
//   lane L: col n = nt*16 + (L&15); k = kt*32 + (L>>4)*16 + j, j=0..15
// ---------------------------------------------------------------------------
__global__ void prep_B_kernel(const float* __restrict__ W,
                              const float* __restrict__ attn,
                              _Float16* __restrict__ Bh) {
    __shared__ float attn_s[CH];
    const int tid = threadIdx.x;
    if (tid < CH) attn_s[tid] = attn[tid];
    __syncthreads();

    float mx = attn_s[0];
    #pragma unroll 4
    for (int k = 1; k < CH; ++k) mx = fmaxf(mx, attn_s[k]);
    float sum = 0.f;
    #pragma unroll 4
    for (int k = 0; k < CH; ++k) sum += __expf(attn_s[k] - mx);
    const float inv = 1.f / sum;

    const int frag = tid >> 5;          // 0..31  == nt*4 + kt
    const int lane = tid & 31;
    const int nt = frag >> 2, kt = frag & 3;
    const int n  = nt * 16 + (lane & 15);
    const int kb = kt * 32 + (lane >> 4) * 16;
    _Float16* dst = Bh + ((size_t)frag * 32 + lane) * 16;
    #pragma unroll
    for (int j = 0; j < 16; ++j) {
        const int k = kb + j;
        const float a = __expf(attn_s[k] - mx) * inv;
        dst[j] = (_Float16)(W[n * CH + k] * a);
    }
}

// ---------------------------------------------------------------------------
// edge_gemm: out[e, :] = relu( (emb[src[e]] + emb[tgt[e]]) @ W'^T + b )
// 128 threads = 4 wave32; each wave = one 16-edge M-tile, all 128 channels.
// Epilogue transposes through LDS so stores are full 512B rows (b128/lane).
// ---------------------------------------------------------------------------
__global__ __launch_bounds__(BLOCK_T)
void edge_gemm_kernel(const float* __restrict__ emb,
                      const long long* __restrict__ eidx,   // [2, E] int64
                      const _Float16* __restrict__ Bh,
                      const float* __restrict__ bias,
                      float* __restrict__ out) {
    __shared__ __attribute__((aligned(32))) _Float16 Bs[32 * 32 * 16]; // 32 KB
    __shared__ __attribute__((aligned(16))) float Cs[WPB][16][CH];     // 32 KB

    // stage all B fragments into LDS (2048 x float4)
    {
        const float4* s = (const float4*)Bh;
        float4*       d = (float4*)Bs;
        for (int i = threadIdx.x; i < 2048; i += BLOCK_T) d[i] = s[i];
    }
    __syncthreads();

    const int wave = threadIdx.x >> 5;
    const int lane = threadIdx.x & 31;
    const int tile = blockIdx.x * WPB + wave;
    if (tile >= N_TILES) return;

    const int e0  = tile * 16;
    const int col = lane & 15;         // edge row within tile / out column
    const int kh  = lane >> 4;         // lane half selects K sub-block
    int e = e0 + col;
    if (e >= N_EDGES) e = N_EDGES - 1; // clamp gathers on tail tile
    const long long si = eidx[e];
    const long long ti = eidx[(size_t)N_EDGES + e];
    const float* __restrict__ srow = emb + (size_t)si * CH;
    const float* __restrict__ trow = emb + (size_t)ti * CH;

    // Build A fragments (contiguous K order, matching B): per lane one
    // 16-elem K-chunk at kt*32 + kh*16.
    v16h A[4];
    #pragma unroll
    for (int kt = 0; kt < 4; ++kt) {
        const int kA = kt * 32 + kh * 16;
        const float4 sa0 = *(const float4*)(srow + kA);
        const float4 sa1 = *(const float4*)(srow + kA + 4);
        const float4 sb0 = *(const float4*)(srow + kA + 8);
        const float4 sb1 = *(const float4*)(srow + kA + 12);
        const float4 ta0 = *(const float4*)(trow + kA);
        const float4 ta1 = *(const float4*)(trow + kA + 4);
        const float4 tb0 = *(const float4*)(trow + kA + 8);
        const float4 tb1 = *(const float4*)(trow + kA + 12);
        v16h a;
        a[0]  = (_Float16)(sa0.x + ta0.x);  a[1]  = (_Float16)(sa0.y + ta0.y);
        a[2]  = (_Float16)(sa0.z + ta0.z);  a[3]  = (_Float16)(sa0.w + ta0.w);
        a[4]  = (_Float16)(sa1.x + ta1.x);  a[5]  = (_Float16)(sa1.y + ta1.y);
        a[6]  = (_Float16)(sa1.z + ta1.z);  a[7]  = (_Float16)(sa1.w + ta1.w);
        a[8]  = (_Float16)(sb0.x + tb0.x);  a[9]  = (_Float16)(sb0.y + tb0.y);
        a[10] = (_Float16)(sb0.z + tb0.z);  a[11] = (_Float16)(sb0.w + tb0.w);
        a[12] = (_Float16)(sb1.x + tb1.x);  a[13] = (_Float16)(sb1.y + tb1.y);
        a[14] = (_Float16)(sb1.z + tb1.z);  a[15] = (_Float16)(sb1.w + tb1.w);
        A[kt] = a;
    }

    // 8 N-tiles x 4 K-steps of v_wmma_f32_16x16x32_f16; bias+ReLU into LDS.
    float (* __restrict__ crow)[CH] = Cs[wave];
    #pragma unroll
    for (int nt = 0; nt < 8; ++nt) {
        v8f c = {};
        #pragma unroll
        for (int kt = 0; kt < 4; ++kt) {
            const v16h b = *(const v16h*)(Bs + ((size_t)(nt * 4 + kt) * 32 + lane) * 16);
            c = __builtin_amdgcn_wmma_f32_16x16x32_f16(
                    /*neg_a=*/false, A[kt], /*neg_b=*/false, b,
                    /*c_mod=*/(short)0, c, /*reuse_a=*/false, /*reuse_b=*/false);
        }
        const float bv = bias[nt * 16 + col];
        // C/D layout: VGPR r -> row (r + kh*8), col = lane&15
        #pragma unroll
        for (int r = 0; r < 8; ++r) {
            const float v = c[r] + bv;
            crow[r + kh * 8][nt * 16 + col] = v > 0.f ? v : 0.f;
        }
    }

    // Store: one full 512B row per instruction (32 lanes x b128).
    // Tail test is wave-uniform -> plain scalar branch, no exec masking.
    float* __restrict__ obase = out + (size_t)e0 * CH;
    if (e0 + 16 <= N_EDGES) {
        #pragma unroll
        for (int rr = 0; rr < 16; ++rr)
            *(float4*)(obase + rr * CH + lane * 4) =
                *(const float4*)&crow[rr][lane * 4];
    } else {
        const int nv = N_EDGES - e0;   // tail tile (8 rows)
        for (int rr = 0; rr < nv; ++rr)
            *(float4*)(obase + rr * CH + lane * 4) =
                *(const float4*)&crow[rr][lane * 4];
    }
}

extern "C" void kernel_launch(void* const* d_in, const int* in_sizes, int n_in,
                              void* d_out, int out_size, void* d_ws, size_t ws_size,
                              hipStream_t stream) {
    const float*     emb  = (const float*)d_in[0];      // [100000, 128] f32
    const long long* eidx = (const long long*)d_in[1];  // [2, 625000] i64
    const float*     attn = (const float*)d_in[2];      // [128] f32
    const float*     W    = (const float*)d_in[3];      // [128, 128] f32
    const float*     bias = (const float*)d_in[4];      // [128] f32
    float*           out  = (float*)d_out;              // [625000, 128] f32
    _Float16*        Bh   = (_Float16*)d_ws;            // 32 KB fragments

    prep_B_kernel<<<1, 1024, 0, stream>>>(W, attn, Bh);
    const int grid = (N_TILES + WPB - 1) / WPB;         // 9766 blocks
    edge_gemm_kernel<<<grid, BLOCK_T, 0, stream>>>(emb, eidx, Bh, bias, out);
}
